// MixedExpertLayer_86199993631253
// MI455X (gfx1250) — matrix operand order repaired and checked by reading, not compiled, not run on hardware
//
#include <hip/hip_runtime.h>
#include <hip/hip_bf16.h>

// ---------------------------------------------------------------------------
// MixedExpertLayer for MI455X (gfx1250, wave32, WMMA bf16)
//   B=4 S=4096 H=1024 I=2048 K_TOP=2 K_CONV=4
// ---------------------------------------------------------------------------
#define B_    4
#define S_    4096
#define H_    1024
#define I_    2048
#define TOK_  (B_ * S_)          // 16384 tokens
#define KPAD  40                 // 32 K-elems + 8 pad (80B rows, 16B aligned)

typedef __attribute__((ext_vector_type(16))) __bf16 v16bf;
typedef __attribute__((ext_vector_type(8)))  float  v8f;

union Frag16 { uint4 q[2]; v16bf v; };
union F4     { float4 v; float f[4]; };
union BF2    { unsigned int u; __bf16 h[2]; };

__device__ __forceinline__ float silu_f(float v) {
  return v / (1.0f + __expf(-v));
}

__device__ __forceinline__ v8f wmma_bf16(v16bf a, v16bf b, v8f c) {
  // D = A(16x32 bf16) * B(32x16 bf16) + C(16x16 f32)
  return __builtin_amdgcn_wmma_f32_16x16x32_bf16(
      /*neg_a=*/false, a, /*neg_b=*/false, b,
      /*c_mod=*/(short)0, c, /*reuse_a=*/false, /*reuse_b=*/false);
}

// A fragment (16x32 bf16, ISA layout): lane L, row = L%16, half = L/16.
// elems 0..7 = K half*8..+7 ; elems 8..15 = K 16+half*8..+7
__device__ __forceinline__ v16bf load_frag_A(const __bf16* rowp, int half) {
  Frag16 f;
  f.q[0] = *(const uint4*)(rowp + half * 8);
  f.q[1] = *(const uint4*)(rowp + 16 + half * 8);
  return f.v;
}
// B fragment (32x16 bf16): lane L, col = L%16, half = L/16.
// elems 0..7 = K half*8..+7 ; elems 8..15 = K 16+half*8..+7 ([N][K] LDS layout)
__device__ __forceinline__ v16bf load_frag_B(const __bf16* rowp, int half) {
  Frag16 f;
  f.q[0] = *(const uint4*)(rowp + half * 8);
  f.q[1] = *(const uint4*)(rowp + 16 + half * 8);
  return f.v;
}

// pack two floats -> one dword of two bf16 (for b32 transposed LDS stores)
__device__ __forceinline__ unsigned int pack_bf2(float a, float b) {
  BF2 p;
  p.h[0] = (__bf16)a;
  p.h[1] = (__bf16)b;
  return p.u;
}

// ---------------------------------------------------------------------------
// Kernel 0: combine-weight table  cw[t][e] = sum_k (idx[t][k]==e)*nw[t][k]
// ---------------------------------------------------------------------------
__global__ __launch_bounds__(256) void cw_kernel(
    const int* __restrict__ topk, const float* __restrict__ nw,
    float* __restrict__ cw) {
  int t = blockIdx.x * 256 + threadIdx.x;   // grid covers TOK_ exactly
  float c0 = 0.f, c1 = 0.f, c2 = 0.f, c3 = 0.f;
#pragma unroll
  for (int k = 0; k < 2; ++k) {
    int e = topk[t * 2 + k];
    float w = nw[t * 2 + k];
    c0 += (e == 0) ? w : 0.f;
    c1 += (e == 1) ? w : 0.f;
    c2 += (e == 2) ? w : 0.f;
    c3 += (e == 3) ? w : 0.f;
  }
  cw[t * 4 + 0] = c0;
  cw[t * 4 + 1] = c1;
  cw[t * 4 + 2] = c2;
  cw[t * 4 + 3] = c3;
}

// ---------------------------------------------------------------------------
// Kernel 1: act_e[t][i] = cw[t][e] * silu(x@Wg_e)[t][i] * (x@Wu_e)[t][i]
// grid: (I/128, TOK/128, 2 experts), block 256 (8 waves, 4Mx2N)
// Software-pipelined: next K-step's global tiles prefetched into registers
// while the current step's WMMAs run.
// ---------------------------------------------------------------------------
__global__ __launch_bounds__(256) void gemm1_kernel(
    const float* __restrict__ x,         // [TOK, H]
    const float* __restrict__ mlp_gate,  // [2, H, I]
    const float* __restrict__ mlp_up,    // [2, H, I]
    const float* __restrict__ cw,        // [TOK, 4]
    __bf16* __restrict__ act) {          // [2, TOK, I] bf16
  __shared__ __align__(16) __bf16 lsA[128 * KPAD];
  __shared__ __align__(16) __bf16 lsBg[128 * KPAD];
  __shared__ __align__(16) __bf16 lsBu[128 * KPAD];

  const int e = blockIdx.z;
  const float* Wg = mlp_gate + (size_t)e * H_ * I_;
  const float* Wu = mlp_up + (size_t)e * H_ * I_;
  __bf16* actE = act + (size_t)e * TOK_ * I_;

  const int tokBase = blockIdx.y * 128;
  const int colBase = blockIdx.x * 128;
  const int tid = threadIdx.x;
  const int lane = tid & 31;
  const int wid = tid >> 5;
  const int wM = (wid & 3) * 32;
  const int wN = (wid >> 2) * 64;
  const int half = lane >> 4;
  const int lr = lane & 15;

  // Per-thread staging coordinates (constant across K-steps)
  // A: 128 rows x 8 float4 = 1024 items, 4 per thread
  int aRow[4], aC4[4];
#pragma unroll
  for (int it = 0; it < 4; ++it) {
    int lin = tid + it * 256;
    aRow[it] = lin >> 3;
    aC4[it] = (lin & 7) * 4;
  }
  // B: 16 k-pairs x 32 n4 = 512 items, 2 per thread (each item = 2 k-rows)
  int bK2[2], bN4[2];
#pragma unroll
  for (int it = 0; it < 2; ++it) {
    int lin = tid + it * 256;
    bK2[it] = (lin >> 5) << 1;       // even k in 0..30
    bN4[it] = (lin & 31) * 4;
  }

  v8f accG[2][4], accU[2][4];
#pragma unroll
  for (int mi = 0; mi < 2; ++mi)
#pragma unroll
    for (int ni = 0; ni < 4; ++ni) {
      accG[mi][ni] = (v8f){0.f, 0.f, 0.f, 0.f, 0.f, 0.f, 0.f, 0.f};
      accU[mi][ni] = (v8f){0.f, 0.f, 0.f, 0.f, 0.f, 0.f, 0.f, 0.f};
    }

  F4 pa[4];            // A prefetch (16 VGPRs)
  F4 pg[2][2], pu[2][2];  // B prefetch (32 VGPRs)

  // ---- prologue: fetch K-step 0
#pragma unroll
  for (int it = 0; it < 4; ++it)
    pa[it].v = *(const float4*)(x + (size_t)(tokBase + aRow[it]) * H_ + aC4[it]);
#pragma unroll
  for (int it = 0; it < 2; ++it) {
    size_t o0 = (size_t)bK2[it] * I_ + colBase + bN4[it];
    pg[it][0].v = *(const float4*)(Wg + o0);
    pg[it][1].v = *(const float4*)(Wg + o0 + I_);
    pu[it][0].v = *(const float4*)(Wu + o0);
    pu[it][1].v = *(const float4*)(Wu + o0 + I_);
  }

  for (int k0 = 0; k0 < H_; k0 += 32) {
    // ---- commit prefetched tiles to LDS
#pragma unroll
    for (int it = 0; it < 4; ++it) {
      __bf16* d = &lsA[aRow[it] * KPAD + aC4[it]];
      d[0] = (__bf16)pa[it].f[0];
      d[1] = (__bf16)pa[it].f[1];
      d[2] = (__bf16)pa[it].f[2];
      d[3] = (__bf16)pa[it].f[3];
    }
#pragma unroll
    for (int it = 0; it < 2; ++it) {
#pragma unroll
      for (int j = 0; j < 4; ++j) {
        *(unsigned int*)&lsBg[(bN4[it] + j) * KPAD + bK2[it]] =
            pack_bf2(pg[it][0].f[j], pg[it][1].f[j]);
        *(unsigned int*)&lsBu[(bN4[it] + j) * KPAD + bK2[it]] =
            pack_bf2(pu[it][0].f[j], pu[it][1].f[j]);
      }
    }
    __syncthreads();

    // ---- issue next K-step's global loads (overlap with WMMAs below)
    int k1 = k0 + 32;
    if (k1 < H_) {
#pragma unroll
      for (int it = 0; it < 4; ++it)
        pa[it].v =
            *(const float4*)(x + (size_t)(tokBase + aRow[it]) * H_ + k1 + aC4[it]);
#pragma unroll
      for (int it = 0; it < 2; ++it) {
        size_t o0 = (size_t)(k1 + bK2[it]) * I_ + colBase + bN4[it];
        pg[it][0].v = *(const float4*)(Wg + o0);
        pg[it][1].v = *(const float4*)(Wg + o0 + I_);
        pu[it][0].v = *(const float4*)(Wu + o0);
        pu[it][1].v = *(const float4*)(Wu + o0 + I_);
      }
    }

    // ---- WMMA compute on current tiles
    v16bf a[2];
#pragma unroll
    for (int mi = 0; mi < 2; ++mi)
      a[mi] = load_frag_A(&lsA[(wM + mi * 16 + lr) * KPAD], half);

#pragma unroll
    for (int ni = 0; ni < 4; ++ni) {
      v16bf bg = load_frag_B(&lsBg[(wN + ni * 16 + lr) * KPAD], half);
#pragma unroll
      for (int mi = 0; mi < 2; ++mi)
        accG[mi][ni] = wmma_bf16(a[mi], bg, accG[mi][ni]);
      v16bf bu = load_frag_B(&lsBu[(wN + ni * 16 + lr) * KPAD], half);
#pragma unroll
      for (int mi = 0; mi < 2; ++mi)
        accU[mi][ni] = wmma_bf16(a[mi], bu, accU[mi][ni]);
    }
    __syncthreads();
  }

  // ---- epilogue: act = cw * silu(g) * u  (bf16 store)
#pragma unroll
  for (int mi = 0; mi < 2; ++mi) {
    int rowBase = tokBase + wM + mi * 16 + half * 8;   // m = v + 8*half
    float cw8[8];
#pragma unroll
    for (int v = 0; v < 8; ++v) cw8[v] = cw[(rowBase + v) * 4 + e];
#pragma unroll
    for (int ni = 0; ni < 4; ++ni) {
      int col = colBase + wN + ni * 16 + lr;
#pragma unroll
      for (int v = 0; v < 8; ++v) {
        float g = accG[mi][ni][v];
        float u = accU[mi][ni][v];
        float r = silu_f(g) * u * cw8[v];
        actE[(size_t)(rowBase + v) * I_ + col] = (__bf16)r;
      }
    }
  }
}

// ---------------------------------------------------------------------------
// Kernel 2: out[t][h] = sum_e act_e[t]@Wd_e[:,h] + conv experts + combine
// grid: (H/128, TOK/128), block 256
// ---------------------------------------------------------------------------
__global__ __launch_bounds__(256) void gemm2_kernel(
    const float* __restrict__ x,         // [TOK, H]
    const float* __restrict__ mlp_down,  // [2, I, H]
    const float* __restrict__ conv_w,    // [2, H, 4]
    const float* __restrict__ cw,        // [TOK, 4]
    const __bf16* __restrict__ act,      // [2, TOK, I]
    float* __restrict__ out) {           // [TOK, H]
  __shared__ __align__(16) __bf16 lsA[128 * KPAD];
  __shared__ __align__(16) __bf16 lsB[128 * KPAD];

  const int tokBase = blockIdx.y * 128;
  const int colBase = blockIdx.x * 128;   // over H
  const int tid = threadIdx.x;
  const int lane = tid & 31;
  const int wid = tid >> 5;
  const int wM = (wid & 3) * 32;
  const int wN = (wid >> 2) * 64;
  const int half = lane >> 4;
  const int lr = lane & 15;

  // staging coordinates
  int aRow[2], aC8[2];
#pragma unroll
  for (int it = 0; it < 2; ++it) {
    int lin = tid + it * 256;          // 0..511
    aRow[it] = lin >> 2;
    aC8[it] = (lin & 3) * 8;
  }
  int bK2[2], bN4[2];
#pragma unroll
  for (int it = 0; it < 2; ++it) {
    int lin = tid + it * 256;
    bK2[it] = (lin >> 5) << 1;
    bN4[it] = (lin & 31) * 4;
  }

  v8f acc[2][4];
#pragma unroll
  for (int mi = 0; mi < 2; ++mi)
#pragma unroll
    for (int ni = 0; ni < 4; ++ni)
      acc[mi][ni] = (v8f){0.f, 0.f, 0.f, 0.f, 0.f, 0.f, 0.f, 0.f};

  for (int e = 0; e < 2; ++e) {
    const __bf16* actE = act + (size_t)e * TOK_ * I_;
    const float* Wd = mlp_down + (size_t)e * I_ * H_;

    uint4 qa[2];
    F4 pb[2][2];
    // prologue fetch for this expert
#pragma unroll
    for (int it = 0; it < 2; ++it)
      qa[it] = *(const uint4*)(actE + (size_t)(tokBase + aRow[it]) * I_ + aC8[it]);
#pragma unroll
    for (int it = 0; it < 2; ++it) {
      size_t o0 = (size_t)bK2[it] * H_ + colBase + bN4[it];
      pb[it][0].v = *(const float4*)(Wd + o0);
      pb[it][1].v = *(const float4*)(Wd + o0 + H_);
    }

    for (int k0 = 0; k0 < I_; k0 += 32) {
      // commit to LDS
#pragma unroll
      for (int it = 0; it < 2; ++it)
        *(uint4*)&lsA[aRow[it] * KPAD + aC8[it]] = qa[it];
#pragma unroll
      for (int it = 0; it < 2; ++it)
#pragma unroll
        for (int j = 0; j < 4; ++j)
          *(unsigned int*)&lsB[(bN4[it] + j) * KPAD + bK2[it]] =
              pack_bf2(pb[it][0].f[j], pb[it][1].f[j]);
      __syncthreads();

      // prefetch next K-step
      int k1 = k0 + 32;
      if (k1 < I_) {
#pragma unroll
        for (int it = 0; it < 2; ++it)
          qa[it] = *(const uint4*)(actE + (size_t)(tokBase + aRow[it]) * I_ +
                                   k1 + aC8[it]);
#pragma unroll
        for (int it = 0; it < 2; ++it) {
          size_t o0 = (size_t)(k1 + bK2[it]) * H_ + colBase + bN4[it];
          pb[it][0].v = *(const float4*)(Wd + o0);
          pb[it][1].v = *(const float4*)(Wd + o0 + H_);
        }
      }

      // WMMA compute
      v16bf a[2];
#pragma unroll
      for (int mi = 0; mi < 2; ++mi)
        a[mi] = load_frag_A(&lsA[(wM + mi * 16 + lr) * KPAD], half);
#pragma unroll
      for (int ni = 0; ni < 4; ++ni) {
        v16bf b = load_frag_B(&lsB[(wN + ni * 16 + lr) * KPAD], half);
#pragma unroll
        for (int mi = 0; mi < 2; ++mi)
          acc[mi][ni] = wmma_bf16(a[mi], b, acc[mi][ni]);
      }
      __syncthreads();
    }
  }

  // ---- epilogue: add depthwise causal conv experts (2,3) + write out
#pragma unroll
  for (int mi = 0; mi < 2; ++mi) {
    int rowBase = tokBase + wM + mi * 16 + half * 8;
    float cw2a[8], cw3a[8];
#pragma unroll
    for (int v = 0; v < 8; ++v) {
      cw2a[v] = cw[(rowBase + v) * 4 + 2];
      cw3a[v] = cw[(rowBase + v) * 4 + 3];
    }
#pragma unroll
    for (int ni = 0; ni < 4; ++ni) {
      int h = colBase + wN + ni * 16 + lr;
      float w2[4], w3[4];
#pragma unroll
      for (int j = 0; j < 4; ++j) {
        w2[j] = conv_w[h * 4 + j];
        w3[j] = conv_w[(size_t)H_ * 4 + h * 4 + j];
      }
#pragma unroll
      for (int v = 0; v < 8; ++v) {
        int t = rowBase + v;
        int b = t >> 12;                 // S_ = 4096
        int s = t & (S_ - 1);
        float c2 = 0.f, c3 = 0.f;
#pragma unroll
        for (int j = 0; j < 4; ++j) {
          int sp = s - 3 + j;
          float xv = (sp >= 0) ? x[((size_t)b * S_ + sp) * H_ + h] : 0.f;
          c2 = fmaf(xv, w2[j], c2);
          c3 = fmaf(xv, w3[j], c3);
        }
        float r = acc[mi][ni][v] + cw2a[v] * silu_f(c2) + cw3a[v] * silu_f(c3);
        out[(size_t)t * H_ + h] = r;
      }
    }
  }
}

// ---------------------------------------------------------------------------
extern "C" void kernel_launch(void* const* d_in, const int* in_sizes, int n_in,
                              void* d_out, int out_size, void* d_ws,
                              size_t ws_size, hipStream_t stream) {
  const float* x        = (const float*)d_in[0];  // [B,S,H]
  const int*   topk     = (const int*)d_in[1];    // [B,S,2]
  const float* nw       = (const float*)d_in[2];  // [B,S,2]
  const float* mlp_gate = (const float*)d_in[3];  // [2,H,I]
  const float* mlp_up   = (const float*)d_in[4];  // [2,H,I]
  const float* mlp_down = (const float*)d_in[5];  // [2,I,H]
  const float* conv_w   = (const float*)d_in[6];  // [2,H,4]
  float* out = (float*)d_out;

  // ws layout: [cw table 256KB][act bf16 2*TOK*I = 128MB]
  float* cw = (float*)d_ws;
  __bf16* act = (__bf16*)((char*)d_ws + (size_t)TOK_ * 4 * sizeof(float));

  cw_kernel<<<TOK_ / 256, 256, 0, stream>>>(topk, nw, cw);
  gemm1_kernel<<<dim3(I_ / 128, TOK_ / 128, 2), 256, 0, stream>>>(
      x, mlp_gate, mlp_up, cw, act);
  gemm2_kernel<<<dim3(H_ / 128, TOK_ / 128), 256, 0, stream>>>(
      x, mlp_down, conv_w, cw, act, out);
}
